// OctopusClassifier_11158325035044
// MI455X (gfx1250) — matrix operand order repaired and checked by reading, not compiled, not run on hardware
//
#include <hip/hip_runtime.h>
#include <hip/hip_bf16.h>
#include <math.h>

// Problem constants (match reference)
#define Dm    4096
#define NHh   8
#define DHh   512            // Dm / NHh
#define DFFf  1024
#define NLl   2
#define Bb    4
#define Tt    16
#define Pp    9              // 1 + Q
#define Ll    25             // P + T
#define Rr    16             // number of mask variants (= T)
#define MROWS (Rr*Bb*Ll)     // 1600
#define EPSC  1e-5f
#define SCALEC 0.04419417382415922f   // 1/sqrt(512)

typedef __attribute__((ext_vector_type(16))) __bf16 v16bf;
typedef __attribute__((ext_vector_type(8)))  float  v8f;

union Frag16 {
    v16bf v;
    unsigned short s[16];
    uint4 q[2];
};

__device__ __forceinline__ unsigned short f2bf(float f) {
    unsigned int u = __float_as_uint(f);
    unsigned int r = u + 0x7FFFu + ((u >> 16) & 1u);   // round-to-nearest-even
    return (unsigned short)(r >> 16);
}

// ---------------------------------------------------------------------------
// fp32 -> bf16 conversion (each element converted exactly once per call)
// n must be a multiple of 4 (true for every tensor here).
// ---------------------------------------------------------------------------
__global__ __launch_bounds__(256) void cvt_bf16_kernel(
    const float* __restrict__ src, unsigned short* __restrict__ dst, int n)
{
    const int i = (blockIdx.x * 256 + threadIdx.x) * 4;
    if (i + 3 < n) {
        const float4 v = *reinterpret_cast<const float4*>(src + i);
        const unsigned int lo = (unsigned int)f2bf(v.x) | ((unsigned int)f2bf(v.y) << 16);
        const unsigned int hi = (unsigned int)f2bf(v.z) | ((unsigned int)f2bf(v.w) << 16);
        *reinterpret_cast<uint2*>(dst + i) = make_uint2(lo, hi);
    }
}

// ---------------------------------------------------------------------------
// Generic GEMM: C[M,N] = act( A[M,K] * W[N,K]^T + bias[N] )
// A, W already bf16 in memory; fp32 WMMA accumulation. Block tile 64x128,
// K-step 32; 8 waves, wave tile 16x64 -> 4 WMMAs/K-step sharing one A frag.
// Software-pipelined global->LDS staging (pure b128 copies, no VALU cvt).
// Grid: x = M-tile (fast) so consecutive blocks reuse the same weight slab
// in L2. Requires M%64==0, N%128==0, K%32==0 (all calls do).
// act: 0=none, 1=relu, 2=gelu(exact erf).  outBf: 1 -> store bf16 instead.
// ---------------------------------------------------------------------------
__global__ __launch_bounds__(256) void wmma_gemm_kernel(
    const unsigned short* __restrict__ A, const unsigned short* __restrict__ W,
    const float* __restrict__ bias, float* __restrict__ Cout,
    int M, int N, int K, int act, int outBf)
{
    __shared__ unsigned short As[64][40];    // 64x32 bf16 (+pad)
    __shared__ unsigned short Ws[128][40];   // 128x32 bf16 (+pad)

    const int tid  = threadIdx.x;
    const int lane = tid & 31;
    const int wave = tid >> 5;
    const int mBase = blockIdx.x * 64;
    const int nBase = blockIdx.y * 128;
    const int wm = (wave & 3) * 16;          // wave M offset in tile
    const int wn = (wave >> 2) * 64;         // wave N offset in tile
    const int hl = lane >> 4;                // lane half (0/1)
    const int lm = lane & 15;

    // A loader: 64 rows x 32 cols, 8 bf16/thread (one b128)
    const int aRow = tid >> 2;
    const int aCol = (tid & 3) * 8;
    // W loader: 128 rows x 32 cols, 16 bf16/thread (two b128)
    const int wRow = tid >> 1;
    const int wCol = (tid & 1) * 16;

    const unsigned short* aPtr = A + (size_t)(mBase + aRow) * K + aCol;
    const unsigned short* wPtr = W + (size_t)(nBase + wRow) * K + wCol;

    // Stage first K-tile into registers
    uint4 aS  = *reinterpret_cast<const uint4*>(aPtr);
    uint4 wS0 = *reinterpret_cast<const uint4*>(wPtr);
    uint4 wS1 = *reinterpret_cast<const uint4*>(wPtr + 8);

    v8f acc0 = {}, acc1 = {}, acc2 = {}, acc3 = {};

    for (int k0 = 0; k0 < K; k0 += 32) {
        // Commit staged registers to LDS (raw bf16, no conversion)
        *reinterpret_cast<uint4*>(&As[aRow][aCol])     = aS;
        *reinterpret_cast<uint4*>(&Ws[wRow][wCol])     = wS0;
        *reinterpret_cast<uint4*>(&Ws[wRow][wCol + 8]) = wS1;
        __syncthreads();

        // Issue next K-tile's global loads early (overlap with WMMA work)
        if (k0 + 32 < K) {
            aS  = *reinterpret_cast<const uint4*>(aPtr + k0 + 32);
            wS0 = *reinterpret_cast<const uint4*>(wPtr + k0 + 32);
            wS1 = *reinterpret_cast<const uint4*>(wPtr + k0 + 40);
        }

        // A fragment: 16x32 bf16, contiguous k order (k = 16*half + elem).
        // lanes 0-15 (row m): elems 0..15 -> K 0..15
        // lanes 16-31 (row m): elems 0..15 -> K 16..31
        Frag16 af, b0, b1, b2, b3;
        af.q[0] = *reinterpret_cast<const uint4*>(&As[wm + lm][hl * 16]);
        af.q[1] = *reinterpret_cast<const uint4*>(&As[wm + lm][hl * 16 + 8]);
        // B fragment (32x16 = W tile column): lane n holds K kbase..kbase+15,
        // kbase = 16*(lane>=16). B[k][n] == W[n][k].
        b0.q[0] = *reinterpret_cast<const uint4*>(&Ws[wn      + lm][hl * 16]);
        b0.q[1] = *reinterpret_cast<const uint4*>(&Ws[wn      + lm][hl * 16 + 8]);
        b1.q[0] = *reinterpret_cast<const uint4*>(&Ws[wn + 16 + lm][hl * 16]);
        b1.q[1] = *reinterpret_cast<const uint4*>(&Ws[wn + 16 + lm][hl * 16 + 8]);
        b2.q[0] = *reinterpret_cast<const uint4*>(&Ws[wn + 32 + lm][hl * 16]);
        b2.q[1] = *reinterpret_cast<const uint4*>(&Ws[wn + 32 + lm][hl * 16 + 8]);
        b3.q[0] = *reinterpret_cast<const uint4*>(&Ws[wn + 48 + lm][hl * 16]);
        b3.q[1] = *reinterpret_cast<const uint4*>(&Ws[wn + 48 + lm][hl * 16 + 8]);

        acc0 = __builtin_amdgcn_wmma_f32_16x16x32_bf16(
            false, af.v, false, b0.v, (short)0, acc0, false, false);
        acc1 = __builtin_amdgcn_wmma_f32_16x16x32_bf16(
            false, af.v, false, b1.v, (short)0, acc1, false, false);
        acc2 = __builtin_amdgcn_wmma_f32_16x16x32_bf16(
            false, af.v, false, b2.v, (short)0, acc2, false, false);
        acc3 = __builtin_amdgcn_wmma_f32_16x16x32_bf16(
            false, af.v, false, b3.v, (short)0, acc3, false, false);
        __syncthreads();
    }

    // Epilogue. C/D layout: VGPR j -> M = j + 8*(lane>=16), N = lane&15.
    const int rBase = mBase + wm + hl * 8;
    const int c0 = nBase + wn + lm;
    const float bia0 = bias[c0];
    const float bia1 = bias[c0 + 16];
    const float bia2 = bias[c0 + 32];
    const float bia3 = bias[c0 + 48];
    auto actf = [&](float v) -> float {
        if (act == 1) return fmaxf(v, 0.f);
        if (act == 2) return 0.5f * v * (1.f + erff(v * 0.70710678118654752f));
        return v;
    };
    unsigned short* CoutB = reinterpret_cast<unsigned short*>(Cout);
    #pragma unroll
    for (int j = 0; j < 8; ++j) {
        const size_t gr = (size_t)(rBase + j) * N;
        const float v0 = actf(acc0[j] + bia0);
        const float v1 = actf(acc1[j] + bia1);
        const float v2 = actf(acc2[j] + bia2);
        const float v3 = actf(acc3[j] + bia3);
        if (outBf) {
            CoutB[gr + c0]      = f2bf(v0);
            CoutB[gr + c0 + 16] = f2bf(v1);
            CoutB[gr + c0 + 32] = f2bf(v2);
            CoutB[gr + c0 + 48] = f2bf(v3);
        } else {
            Cout[gr + c0]      = v0;
            Cout[gr + c0 + 16] = v1;
            Cout[gr + c0 + 32] = v2;
            Cout[gr + c0 + 48] = v3;
        }
    }
}

// ---------------------------------------------------------------------------
// Build combined sequence replicated across the 16 mask variants.
// Writes fp32 master X and bf16 copy Xb.
// ---------------------------------------------------------------------------
__global__ __launch_bounds__(256) void assemble_kernel(
    float* __restrict__ X, unsigned short* __restrict__ Xb,
    const float* __restrict__ xp, const float* __restrict__ cls,
    const float* __restrict__ qry)
{
    const int row = blockIdx.x;         // 0..MROWS-1
    const int l  = row % Ll;
    const int rb = row / Ll;
    const int b  = rb % Bb;
    const float* src;
    if (l == 0)        src = cls;
    else if (l < Pp)   src = qry + (size_t)(l - 1) * Dm;
    else               src = xp  + (size_t)(b * Tt + (l - Pp)) * Dm;
    float* dst = X + (size_t)row * Dm;
    unsigned short* dstb = Xb + (size_t)row * Dm;
    for (int i = threadIdx.x; i < Dm; i += 256) {
        const float v = src[i];
        dst[i] = v;
        dstb[i] = f2bf(v);
    }
}

// ---------------------------------------------------------------------------
// Masked attention. One block per (r,b,l) row, one wave per head.
// QKV row layout: [q(D) | k(D) | v(D)], each D is NH x DH (h-major).
// Output written directly as bf16 (only consumed by the Wo GEMM).
// ---------------------------------------------------------------------------
__global__ __launch_bounds__(256) void attn_kernel(
    const float* __restrict__ QKV, unsigned short* __restrict__ O)
{
    const int row  = blockIdx.x;
    const int tid  = threadIdx.x;
    const int lane = tid & 31;
    const int h    = tid >> 5;                 // head (8 waves)
    const int r    = row / (Bb * Ll);
    const int l    = row % Ll;
    const int rowbase = row - l;               // start of this (r,b) sequence
    int nvalid = Pp + 1 + r;
    if (nvalid > Ll) nvalid = Ll;

    const float* q = QKV + (size_t)row * (3 * Dm) + h * DHh;
    float qv[16];
    #pragma unroll
    for (int i = 0; i < 16; ++i) qv[i] = q[lane + 32 * i];

    // scores: lane m ends up holding score for key m
    float myscore = -1e30f;
    for (int m = 0; m < nvalid; ++m) {
        const float* kp = QKV + (size_t)(rowbase + m) * (3 * Dm) + Dm + h * DHh;
        float s = 0.f;
        #pragma unroll
        for (int i = 0; i < 16; ++i) s += qv[i] * kp[lane + 32 * i];
        #pragma unroll
        for (int off = 16; off; off >>= 1) s += __shfl_xor(s, off, 32);
        if (lane == m) myscore = s * SCALEC;
    }
    float mx = myscore;
    #pragma unroll
    for (int off = 16; off; off >>= 1) mx = fmaxf(mx, __shfl_xor(mx, off, 32));
    float p = (lane < nvalid) ? expf(myscore - mx) : 0.f;
    float ssum = p;
    #pragma unroll
    for (int off = 16; off; off >>= 1) ssum += __shfl_xor(ssum, off, 32);
    p /= ssum;

    float o[16];
    #pragma unroll
    for (int i = 0; i < 16; ++i) o[i] = 0.f;
    for (int m = 0; m < nvalid; ++m) {
        const float pm = __shfl(p, m, 32);
        const float* vp = QKV + (size_t)(rowbase + m) * (3 * Dm) + 2 * Dm + h * DHh;
        #pragma unroll
        for (int i = 0; i < 16; ++i) o[i] += pm * vp[lane + 32 * i];
    }
    unsigned short* op = O + (size_t)row * Dm + h * DHh;
    #pragma unroll
    for (int i = 0; i < 16; ++i) op[lane + 32 * i] = f2bf(o[i]);
}

// ---------------------------------------------------------------------------
// X = LayerNorm(X + Y) * g + b, in place; also writes bf16 copy Xb.
// ---------------------------------------------------------------------------
__global__ __launch_bounds__(256) void add_ln_kernel(
    float* __restrict__ X, unsigned short* __restrict__ Xb,
    const float* __restrict__ Y,
    const float* __restrict__ g, const float* __restrict__ bt)
{
    __shared__ float red[256];
    const int row = blockIdx.x;
    const int tid = threadIdx.x;
    float v[16];
    float s = 0.f;
    #pragma unroll
    for (int i = 0; i < 16; ++i) {
        const int idx = tid + 256 * i;
        v[i] = X[(size_t)row * Dm + idx] + Y[(size_t)row * Dm + idx];
        s += v[i];
    }
    red[tid] = s; __syncthreads();
    for (int st = 128; st > 0; st >>= 1) {
        if (tid < st) red[tid] += red[tid + st];
        __syncthreads();
    }
    const float mean = red[0] * (1.f / Dm);
    __syncthreads();
    s = 0.f;
    #pragma unroll
    for (int i = 0; i < 16; ++i) { const float d = v[i] - mean; s += d * d; }
    red[tid] = s; __syncthreads();
    for (int st = 128; st > 0; st >>= 1) {
        if (tid < st) red[tid] += red[tid + st];
        __syncthreads();
    }
    const float rstd = rsqrtf(red[0] * (1.f / Dm) + EPSC);
    #pragma unroll
    for (int i = 0; i < 16; ++i) {
        const int idx = tid + 256 * i;
        const float o = g[idx] * (v[i] - mean) * rstd + bt[idx];
        X[(size_t)row * Dm + idx]  = o;
        Xb[(size_t)row * Dm + idx] = f2bf(o);
    }
}

// CLS row gather (bf16): CLS[r*B+b][:] = Xb[(r*B+b)*L + 0][:]
__global__ __launch_bounds__(256) void extract_cls_kernel(
    const unsigned short* __restrict__ Xb, unsigned short* __restrict__ CLS)
{
    const int rb = blockIdx.x;     // 0..63
    const unsigned short* src = Xb + (size_t)rb * Ll * Dm;
    unsigned short* dst = CLS + (size_t)rb * Dm;
    for (int i = threadIdx.x; i < Dm; i += 256) dst[i] = src[i];
}

// Final head: logits[b][t][c] = H[t*B+b] . Wm2[c] + bm2[c]
__global__ __launch_bounds__(128) void logits_kernel(
    const float* __restrict__ H, const float* __restrict__ Wm2,
    const float* __restrict__ bm2, float* __restrict__ out)
{
    const int rb   = blockIdx.x;       // r*B + b, 0..63
    const int tid  = threadIdx.x;
    const int lane = tid & 31;
    const int c    = tid >> 5;         // 4 waves -> 4 classes
    const int r = rb / Bb, b = rb % Bb;
    const float* hp = H   + (size_t)rb * (Dm / 2);
    const float* wp = Wm2 + (size_t)c  * (Dm / 2);
    float s = 0.f;
    for (int i = lane; i < Dm / 2; i += 32) s += hp[i] * wp[i];
    #pragma unroll
    for (int off = 16; off; off >>= 1) s += __shfl_xor(s, off, 32);
    if (lane == 0) out[(size_t)b * (Tt * 4) + r * 4 + c] = s + bm2[c];
}

// ---------------------------------------------------------------------------
extern "C" void kernel_launch(void* const* d_in, const int* in_sizes, int n_in,
                              void* d_out, int out_size, void* d_ws, size_t ws_size,
                              hipStream_t stream)
{
    (void)in_sizes; (void)n_in; (void)out_size; (void)ws_size;
    // Input order: hidden_states, queries, cls, Wp, bp,
    //   per layer (x2): Wqkv,bqkv,Wo,bo,ln1g,ln1b,ln2g,ln2b,W1,b1,W2,b2,
    //   Wm1,bm1,Wm2,bm2
    const float* hs  = (const float*)d_in[0];
    const float* qry = (const float*)d_in[1];
    const float* cls = (const float*)d_in[2];
    const float* Wp  = (const float*)d_in[3];
    const float* bp  = (const float*)d_in[4];
    const float* Wm1 = (const float*)d_in[29];
    const float* bm1 = (const float*)d_in[30];
    const float* Wm2 = (const float*)d_in[31];
    const float* bm2 = (const float*)d_in[32];

    char* wsp = (char*)d_ws;
    auto alloc = [&](size_t bytes) {
        char* p = wsp;
        wsp += (bytes + 255) & ~(size_t)255;
        return p;
    };
    typedef unsigned short u16;
    // bf16 weight copies (converted once per call)
    u16* Wpb  = (u16*)alloc((size_t)Dm * Dm * 2);
    u16* Wm1b = (u16*)alloc((size_t)(Dm / 2) * Dm * 2);
    u16* Wqkvb[NLl]; u16* Wob[NLl]; u16* W1b[NLl]; u16* W2b[NLl];
    for (int l = 0; l < NLl; ++l) {
        Wqkvb[l] = (u16*)alloc((size_t)3 * Dm * Dm * 2);
        Wob[l]   = (u16*)alloc((size_t)Dm * Dm * 2);
        W1b[l]   = (u16*)alloc((size_t)DFFf * Dm * 2);
        W2b[l]   = (u16*)alloc((size_t)Dm * DFFf * 2);
    }
    u16*   hsb  = (u16*)alloc((size_t)Bb * Tt * Dm * 2);
    float* X    = (float*)alloc((size_t)MROWS * Dm * 4);       // fp32 master
    u16*   Xb   = (u16*)alloc((size_t)MROWS * Dm * 2);         // bf16 copy
    float* QKV  = (float*)alloc((size_t)MROWS * 3 * Dm * 4);   // qkv / tmp
    u16*   Obb  = (u16*)alloc((size_t)MROWS * Dm * 2);         // attn out bf16
    u16*   Fbb  = (u16*)alloc((size_t)MROWS * DFFf * 2);       // ffn hidden bf16
    float* xp   = (float*)alloc((size_t)Bb * Tt * Dm * 4);
    u16*   CLSb = (u16*)alloc((size_t)Rr * Bb * Dm * 2);
    float* Hb   = (float*)alloc((size_t)Rr * Bb * (Dm / 2) * 4);

    const dim3 blk(256);
    auto cvt = [&](const float* s, u16* d, size_t n) {
        cvt_bf16_kernel<<<dim3((unsigned)((n / 4 + 255) / 256)), blk, 0, stream>>>(s, d, (int)n);
    };

    // Convert all weights to bf16 once
    cvt(Wp,  Wpb,  (size_t)Dm * Dm);
    cvt(Wm1, Wm1b, (size_t)(Dm / 2) * Dm);
    cvt(hs,  hsb,  (size_t)Bb * Tt * Dm);
    for (int l = 0; l < NLl; ++l) {
        cvt((const float*)d_in[5 + 12 * l + 0],  Wqkvb[l], (size_t)3 * Dm * Dm);
        cvt((const float*)d_in[5 + 12 * l + 2],  Wob[l],   (size_t)Dm * Dm);
        cvt((const float*)d_in[5 + 12 * l + 8],  W1b[l],   (size_t)DFFf * Dm);
        cvt((const float*)d_in[5 + 12 * l + 10], W2b[l],   (size_t)Dm * DFFf);
    }

    // xp = hs @ Wp^T + bp          (M=64, N=4096, K=4096)
    wmma_gemm_kernel<<<dim3((Bb * Tt) / 64, Dm / 128), blk, 0, stream>>>(
        hsb, Wpb, bp, xp, Bb * Tt, Dm, Dm, 0, 0);
    // Replicate prefix + context across the 16 mask variants
    assemble_kernel<<<MROWS, blk, 0, stream>>>(X, Xb, xp, cls, qry);

    for (int l = 0; l < NLl; ++l) {
        const float* bqkv = (const float*)d_in[5 + 12 * l + 1];
        const float* bo   = (const float*)d_in[5 + 12 * l + 3];
        const float* ln1g = (const float*)d_in[5 + 12 * l + 4];
        const float* ln1b = (const float*)d_in[5 + 12 * l + 5];
        const float* ln2g = (const float*)d_in[5 + 12 * l + 6];
        const float* ln2b = (const float*)d_in[5 + 12 * l + 7];
        const float* b1   = (const float*)d_in[5 + 12 * l + 9];
        const float* b2   = (const float*)d_in[5 + 12 * l + 11];

        // QKV = X @ Wqkv^T + bqkv   (1600 x 12288 x 4096)
        wmma_gemm_kernel<<<dim3(MROWS / 64, 3 * Dm / 128), blk, 0, stream>>>(
            Xb, Wqkvb[l], bqkv, QKV, MROWS, 3 * Dm, Dm, 0, 0);
        attn_kernel<<<MROWS, blk, 0, stream>>>(QKV, Obb);
        // Output projection (reuse QKV buffer as fp32 temp)
        wmma_gemm_kernel<<<dim3(MROWS / 64, Dm / 128), blk, 0, stream>>>(
            Obb, Wob[l], bo, QKV, MROWS, Dm, Dm, 0, 0);
        add_ln_kernel<<<MROWS, blk, 0, stream>>>(X, Xb, QKV, ln1g, ln1b);
        // FFN: hidden written directly as bf16 (+ReLU)
        wmma_gemm_kernel<<<dim3(MROWS / 64, DFFf / 128), blk, 0, stream>>>(
            Xb, W1b[l], b1, (float*)Fbb, MROWS, DFFf, Dm, 1, 1);
        wmma_gemm_kernel<<<dim3(MROWS / 64, Dm / 128), blk, 0, stream>>>(
            Fbb, W2b[l], b2, QKV, MROWS, Dm, DFFf, 0, 0);
        add_ln_kernel<<<MROWS, blk, 0, stream>>>(X, Xb, QKV, ln2g, ln2b);
    }

    extract_cls_kernel<<<Rr * Bb, blk, 0, stream>>>(Xb, CLSb);
    // H = gelu(CLS @ Wm1^T + bm1)   (64 x 2048 x 4096)
    wmma_gemm_kernel<<<dim3((Rr * Bb) / 64, (Dm / 2) / 128), blk, 0, stream>>>(
        CLSb, Wm1b, bm1, Hb, Rr * Bb, Dm / 2, Dm, 2, 0);
    logits_kernel<<<Rr * Bb, dim3(128), 0, stream>>>(Hb, Wm2, bm2, (float*)d_out);
}
